// BiDAF_81080392614904
// MI455X (gfx1250) — matrix operand order repaired and checked by reading, not compiled, not run on hardware
//
#include <hip/hip_runtime.h>
#include <hip/hip_bf16.h>

typedef __attribute__((ext_vector_type(16))) _Float16 v16h;
typedef __attribute__((ext_vector_type(8)))  _Float16 v8h;
typedef __attribute__((ext_vector_type(8)))  float    v8f;

#define B_   8
#define LC   256
#define LQ   32
#define CLEN 16
#define EMB  364
#define EMBP 384   // EMB padded to multiple of 32
#define EMBN 368   // EMB padded to multiple of 16 (N dim of highway weights)

static __device__ __forceinline__ float sigf(float x) { return 1.0f / (1.0f + expf(-x)); }

// ---------------------------------------------------------------------------
// Tiled WMMA GEMM, fully unguarded loads:
//   D[m,n] = sum_k A[m,k] * B'[k,n] (+ bias[n])
// A: f16, (M x lda), M % 16 == 0, K % 32 == 0, lda % 32 == 0 (padded, zeros)
// transB=1: B = W (Npad x ldb) f16 row-major, padded with zero rows/cols
// transB=0: B = (K x ldb) f16 row-major
// Only the 8 output stores are guarded (col < N).  gridDim.z = batch.
// ---------------------------------------------------------------------------
__global__ void gemm_wmma(const _Float16* __restrict__ A, int lda, long sA,
                          const _Float16* __restrict__ Bm, int ldb, long sB,
                          const float* __restrict__ bias,
                          float* __restrict__ D, int ldd, long sD,
                          int M, int N, int K, int transB)
{
    int z = blockIdx.z;
    A  += (long)z * sA;
    Bm += (long)z * sB;
    D  += (long)z * sD;

    int wave  = threadIdx.x >> 5;
    int lane  = threadIdx.x & 31;
    int lmod  = lane & 15;
    int lhalf = lane >> 4;

    int tilesN = (N + 15) >> 4;
    int tilesM = M >> 4;
    int gw = blockIdx.x * (blockDim.x >> 5) + wave;
    if (gw >= tilesM * tilesN) return;
    int tm = gw / tilesN, tn = gw % tilesN;

    const _Float16* arow = A + (long)(tm * 16 + lmod) * lda;
    int colB = tn * 16 + lmod;

    v8f acc = {};
    if (transB) {
        const _Float16* brow = Bm + (long)colB * ldb;
        for (int k0 = 0; k0 < K; k0 += 32) {
            v16h a, b;
            v8h a0 = *(const v8h*)(arow + k0 + lhalf * 16);
            v8h a1 = *(const v8h*)(arow + k0 + lhalf * 16 + 8);
            v8h b0 = *(const v8h*)(brow + k0 + lhalf * 16);
            v8h b1 = *(const v8h*)(brow + k0 + lhalf * 16 + 8);
#pragma unroll
            for (int i = 0; i < 8; ++i) {
                a[i] = a0[i]; a[i + 8] = a1[i];
                b[i] = b0[i]; b[i + 8] = b1[i];
            }
            acc = __builtin_amdgcn_wmma_f32_16x16x32_f16(false, a, false, b,
                                                         (short)0, acc, false, false);
        }
    } else {
        for (int k0 = 0; k0 < K; k0 += 32) {
            v16h a, b;
            v8h a0 = *(const v8h*)(arow + k0 + lhalf * 16);
            v8h a1 = *(const v8h*)(arow + k0 + lhalf * 16 + 8);
#pragma unroll
            for (int i = 0; i < 8; ++i) { a[i] = a0[i]; a[i + 8] = a1[i]; }
#pragma unroll
            for (int j = 0; j < 16; ++j)
                b[j] = Bm[(long)(k0 + lhalf * 16 + j) * ldb + colB];
            acc = __builtin_amdgcn_wmma_f32_16x16x32_f16(false, a, false, b,
                                                         (short)0, acc, false, false);
        }
    }
#pragma unroll
    for (int r = 0; r < 8; ++r) {
        int row = tm * 16 + lhalf * 8 + r;
        int col = tn * 16 + lmod;
        if (col < N)
            D[(long)row * ldd + col] = acc[r] + (bias ? bias[col] : 0.f);
    }
}

// ---------------------------------------------------------------------------
// Persistent LSTM scan: one workgroup (1024 thr = 32 waves) runs all T steps.
// h (f16, 16 rows: 8 live + 8 permanently-zero pad rows -> unguarded A loads)
// and c (f32) live in LDS.  Per step: g = h @ Whh^T via WMMA into an LDS gate
// buffer, then fused sigmoid/tanh gate math.  pre is (B,T,4H) with bias folded.
// Dynamic LDS: 192*H bytes.
// ---------------------------------------------------------------------------
__global__ void __launch_bounds__(1024) lstm_scan(
    const float* __restrict__ pre, const _Float16* __restrict__ Whh,
    float* __restrict__ out, int H, int T, int reverse, int outOff, int outLD)
{
    extern __shared__ char smem[];
    _Float16* hbuf = (_Float16*)smem;               // 16*H f16 (rows 8..15 zero)
    float*    cbuf = (float*)(smem + 16 * H * 2);   // 8*H f32
    float*    gbuf = cbuf + 8 * H;                  // 8*4H f32

    int tid   = threadIdx.x;
    int wave  = tid >> 5;
    int lane  = tid & 31;
    int lmod  = lane & 15;
    int lhalf = lane >> 4;
    int H4    = 4 * H;
    int ntiles = H4 >> 4;

    for (int e = tid; e < 16 * H; e += 1024) hbuf[e] = (_Float16)0.f;
    for (int e = tid; e < 8 * H; e += 1024)  cbuf[e] = 0.f;
    __syncthreads();

    for (int step = 0; step < T; ++step) {
        int t = reverse ? (T - 1 - step) : step;

        // ---- g = h @ Whh^T ----
        const _Float16* hrow = hbuf + lmod * H;
        for (int tile = wave; tile < ntiles; tile += 32) {
            v8f acc = {};
            const _Float16* wrow = Whh + (long)(tile * 16 + lmod) * H;
            for (int k0 = 0; k0 < H; k0 += 32) {
                v16h a, b;
                v8h a0 = *(const v8h*)(hrow + k0 + lhalf * 16);
                v8h a1 = *(const v8h*)(hrow + k0 + lhalf * 16 + 8);
                v8h b0 = *(const v8h*)(wrow + k0 + lhalf * 16);
                v8h b1 = *(const v8h*)(wrow + k0 + lhalf * 16 + 8);
#pragma unroll
                for (int i = 0; i < 8; ++i) {
                    a[i] = a0[i]; a[i + 8] = a1[i];
                    b[i] = b0[i]; b[i + 8] = b1[i];
                }
                acc = __builtin_amdgcn_wmma_f32_16x16x32_f16(false, a, false, b,
                                                             (short)0, acc, false, false);
            }
            if (lhalf == 0) {       // rows 0..7 are the 8 batch entries
#pragma unroll
                for (int r = 0; r < 8; ++r)
                    gbuf[r * H4 + tile * 16 + lmod] = acc[r];
            }
        }
        __syncthreads();

        // ---- gates: i,f,g,o ----
        for (int e = tid; e < 8 * H; e += 1024) {
            int bb = e / H, hh = e - bb * H;
            long pb = ((long)bb * T + t) * H4;
            float gi = gbuf[bb * H4 + hh]         + pre[pb + hh];
            float gf = gbuf[bb * H4 + H + hh]     + pre[pb + H + hh];
            float gg = gbuf[bb * H4 + 2 * H + hh] + pre[pb + 2 * H + hh];
            float go = gbuf[bb * H4 + 3 * H + hh] + pre[pb + 3 * H + hh];
            float cc = sigf(gf) * cbuf[e] + sigf(gi) * tanhf(gg);
            float hv = sigf(go) * tanhf(cc);
            cbuf[e] = cc;
            hbuf[bb * H + hh] = (_Float16)hv;
            out[((long)bb * T + t) * outLD + outOff + hh] = hv;
        }
        __syncthreads();
    }
}

// ---------------------------------------------------------------------------
// Embedding: char depthwise 5x5 conv (+pad 2) -> pointwise 64x64 -> relu ->
// max over char dim, concat with word embedding.  One block per (b,l).
// ---------------------------------------------------------------------------
__global__ void embed_kernel(const int* __restrict__ wid, const int* __restrict__ cid,
                             const float* __restrict__ word_mat, const float* __restrict__ char_mat,
                             const float* __restrict__ dw_w, const float* __restrict__ dw_b,
                             const float* __restrict__ pw_w, const float* __restrict__ pw_b,
                             float* __restrict__ x, int L)
{
    __shared__ float dwbuf[64 * CLEN];
    __shared__ float pwbuf[64 * CLEN];
    int b = blockIdx.x / L, l = blockIdx.x % L;
    int tid = threadIdx.x;

    for (int idx = tid; idx < 64 * CLEN; idx += 256) {
        int c = idx >> 4, cl = idx & 15;
        float acc = dw_b[c];
#pragma unroll
        for (int dl = 0; dl < 5; ++dl) {
            int ll = l + dl - 2;
            if (ll < 0 || ll >= L) continue;
#pragma unroll
            for (int dcl = 0; dcl < 5; ++dcl) {
                int cc = cl + dcl - 2;
                if (cc < 0 || cc >= CLEN) continue;
                int id = cid[((long)b * L + ll) * CLEN + cc];
                acc += dw_w[c * 25 + dl * 5 + dcl] * char_mat[(long)id * 64 + c];
            }
        }
        dwbuf[idx] = acc;
    }
    __syncthreads();

    for (int idx = tid; idx < 64 * CLEN; idx += 256) {
        int o = idx >> 4, cl = idx & 15;
        float acc = pw_b[o];
        for (int i = 0; i < 64; ++i) acc += pw_w[o * 64 + i] * dwbuf[i * CLEN + cl];
        pwbuf[idx] = fmaxf(acc, 0.f);
    }
    __syncthreads();

    long rbase = ((long)b * L + l) * EMB;
    if (tid < 64) {
        float mx = pwbuf[tid * CLEN];
        for (int cl = 1; cl < CLEN; ++cl) mx = fmaxf(mx, pwbuf[tid * CLEN + cl]);
        x[rbase + tid] = mx;
    }
    int w = wid[(long)b * L + l];
    for (int j = tid; j < 300; j += 256)
        x[rbase + 64 + j] = word_mat[(long)w * 300 + j];
}

// ---------------------------------------------------------------------------
// Small utility kernels
// ---------------------------------------------------------------------------
__global__ void add2(const float* __restrict__ a, const float* __restrict__ b,
                     float* __restrict__ o, int n)
{
    int i = blockIdx.x * blockDim.x + threadIdx.x;
    if (i < n) o[i] = a[i] + b[i];
}

// f32 (R x K) -> f16 (Rpad x Kpad), zero padded
__global__ void stage_f16(const float* __restrict__ src, int R, int K,
                          _Float16* __restrict__ dst, int Rpad, int Kpad)
{
    int i = blockIdx.x * blockDim.x + threadIdx.x;
    if (i >= Rpad * Kpad) return;
    int r = i / Kpad, c = i - r * Kpad;
    dst[i] = (r < R && c < K) ? (_Float16)src[(long)r * K + c] : (_Float16)0.f;
}

__global__ void cw_mul_f16(const float* __restrict__ C, const float* __restrict__ w,
                           _Float16* __restrict__ o, int n, int d)
{
    int i = blockIdx.x * blockDim.x + threadIdx.x;
    if (i < n) o[i] = (_Float16)(C[i] * w[i % d]);
}

__global__ void highway_comb(float* __restrict__ x, const float* __restrict__ gp,
                             const float* __restrict__ lp, int n)
{
    int i = blockIdx.x * blockDim.x + threadIdx.x;
    if (i < n) {
        float g = sigf(gp[i]);
        x[i] = g * fmaxf(lp[i], 0.f) + (1.f - g) * x[i];
    }
}

// out[row] (+)= dot(A[row,:], w) + bias[0]; one wave per row
__global__ void rowdot(const float* __restrict__ A, int D, const float* __restrict__ w,
                       const float* __restrict__ bias, float* __restrict__ out,
                       int rows, int acc)
{
    int row  = (blockIdx.x * blockDim.x + threadIdx.x) >> 5;
    int lane = threadIdx.x & 31;
    if (row >= rows) return;
    const float* a = A + (long)row * D;
    float s = 0.f;
    for (int d = lane; d < D; d += 32) s += a[d] * w[d];
    for (int off = 16; off > 0; off >>= 1) s += __shfl_down(s, off);
    if (lane == 0) {
        float v = s + bias[0];
        out[row] = acc ? (out[row] + v) : v;
    }
}

// one wave per (b,i): softmax over the 32 q positions -> f16 a, record row max
__global__ void attn_softmax(const float* __restrict__ s, const float* __restrict__ cterm,
                             const float* __restrict__ qterm, const float* __restrict__ bcq,
                             _Float16* __restrict__ a, float* __restrict__ smax)
{
    int row  = (blockIdx.x * blockDim.x + threadIdx.x) >> 5;
    int lane = threadIdx.x & 31;
    if (row >= B_ * LC) return;
    int b = row >> 8;
    float v = s[(long)row * LQ + lane] + cterm[row] + qterm[b * LQ + lane] + bcq[0];
    float mx = v;
    for (int off = 16; off > 0; off >>= 1) mx = fmaxf(mx, __shfl_down(mx, off));
    mx = __shfl(mx, 0);
    float e = expf(v - mx);
    float sum = e;
    for (int off = 16; off > 0; off >>= 1) sum += __shfl_down(sum, off);
    sum = __shfl(sum, 0);
    a[(long)row * LQ + lane] = (_Float16)(e / sum);
    if (lane == 0) smax[row] = mx;
}

// per batch: b = softmax_i(smax); q2c[b,:] = sum_i b_i * C[b,i,:]
__global__ void q2c_kernel(const float* __restrict__ smax, const float* __restrict__ C,
                           float* __restrict__ q2c)
{
    __shared__ float w[LC];
    __shared__ float red[LC];
    int b = blockIdx.x, tid = threadIdx.x;
    float v = smax[b * LC + tid];
    red[tid] = v; __syncthreads();
    for (int s = 128; s > 0; s >>= 1) { if (tid < s) red[tid] = fmaxf(red[tid], red[tid + s]); __syncthreads(); }
    float mx = red[0]; __syncthreads();
    float e = expf(v - mx);
    red[tid] = e; __syncthreads();
    for (int s = 128; s > 0; s >>= 1) { if (tid < s) red[tid] += red[tid + s]; __syncthreads(); }
    float sum = red[0];
    w[tid] = e / sum; __syncthreads();
    for (int d = tid; d < 512; d += 256) {
        float acc = 0.f;
        for (int i = 0; i < LC; ++i) acc += w[i] * C[((long)b * LC + i) * 512 + d];
        q2c[b * 512 + d] = acc;
    }
}

// g = [C, c2q, C*c2q, C*q2c] -> f32 g and f16 gh
__global__ void build_g(const float* __restrict__ C, const float* __restrict__ c2q,
                        const float* __restrict__ q2c, float* __restrict__ g,
                        _Float16* __restrict__ gh, int n)
{
    int idx = blockIdx.x * blockDim.x + threadIdx.x;
    if (idx >= n) return;
    int row = idx >> 11, col = idx & 2047;
    int b = row >> 8;
    float v;
    if (col < 512)        v = C[(long)row * 512 + col];
    else if (col < 1024)  v = c2q[(long)row * 512 + col - 512];
    else if (col < 1536)  v = C[(long)row * 512 + col - 1024] * c2q[(long)row * 512 + col - 1024];
    else                  v = C[(long)row * 512 + col - 1536] * q2c[b * 512 + col - 1536];
    g[idx] = v;
    gh[idx] = (_Float16)v;
}

// mask by (Cwid==0) -> -1e30, then log_softmax over the 256 positions of batch b
__global__ void mask_logsoftmax(const float* __restrict__ p, const int* __restrict__ Cwid,
                                float* __restrict__ outp)
{
    __shared__ float red[LC];
    int b = blockIdx.x, tid = threadIdx.x;
    float v = (Cwid[b * LC + tid] == 0) ? -1e30f : p[b * LC + tid];
    red[tid] = v; __syncthreads();
    for (int s = 128; s > 0; s >>= 1) { if (tid < s) red[tid] = fmaxf(red[tid], red[tid + s]); __syncthreads(); }
    float mx = red[0]; __syncthreads();
    red[tid] = expf(v - mx); __syncthreads();
    for (int s = 128; s > 0; s >>= 1) { if (tid < s) red[tid] += red[tid + s]; __syncthreads(); }
    float lse = logf(red[0]) + mx;
    outp[b * LC + tid] = v - lse;
}

// ---------------------------------------------------------------------------
// Host orchestration
// ---------------------------------------------------------------------------
extern "C" void kernel_launch(void* const* d_in, const int* in_sizes, int n_in,
                              void* d_out, int out_size, void* d_ws, size_t ws_size,
                              hipStream_t stream)
{
    (void)in_sizes; (void)n_in; (void)out_size; (void)ws_size;

    const int*   Cwid     = (const int*)d_in[0];
    const int*   Ccid     = (const int*)d_in[1];
    const int*   Qwid     = (const int*)d_in[2];
    const int*   Qcid     = (const int*)d_in[3];
    const float* word_mat = (const float*)d_in[4];
    const float* char_mat = (const float*)d_in[5];
    const float* dw_w     = (const float*)d_in[6];
    const float* dw_b     = (const float*)d_in[7];
    const float* pw_w     = (const float*)d_in[8];
    const float* pw_b     = (const float*)d_in[9];
    const float* hw_lin_w = (const float*)d_in[10];
    const float* hw_lin_b = (const float*)d_in[11];
    const float* hw_gate_w= (const float*)d_in[12];
    const float* hw_gate_b= (const float*)d_in[13];
    const float* enc_Wih  = (const float*)d_in[14];
    const float* enc_Whh  = (const float*)d_in[15];
    const float* enc_bih  = (const float*)d_in[16];
    const float* enc_bhh  = (const float*)d_in[17];
    const float* att_wc   = (const float*)d_in[18];
    const float* att_bc   = (const float*)d_in[19];
    const float* att_wq   = (const float*)d_in[20];
    const float* att_bq   = (const float*)d_in[21];
    const float* att_wcq  = (const float*)d_in[22];
    const float* att_bcq  = (const float*)d_in[23];
    const int base[6] = {24, 28, 32, 36, 40, 44};   // m1_f m1_b m2_f m2_b out_f out_b

    char* wp = (char*)d_ws;
    auto allocB = [&](size_t bytes) { char* p = wp; wp += (bytes + 255) & ~(size_t)255; return p; };
    auto allocF = [&](size_t n) { return (float*)allocB(n * sizeof(float)); };
    auto allocH = [&](size_t n) { return (_Float16*)allocB(n * sizeof(_Float16)); };
    auto g1 = [](long n) { return dim3((unsigned)((n + 255) / 256)); };

    auto gemm = [&](const _Float16* A, int lda, long sA, const _Float16* W, int ldb, long sB,
                    const float* bias, float* D, int ldd, long sD,
                    int M, int N, int K, int transB, int batch) {
        long waves = (long)(M / 16) * ((N + 15) / 16);
        dim3 grid((unsigned)((waves + 7) / 8), 1, (unsigned)batch);
        gemm_wmma<<<grid, 256, 0, stream>>>(A, lda, sA, W, ldb, sB, bias, D, ldd, sD, M, N, K, transB);
    };
    auto stage = [&](const float* src, int R, int K, _Float16* dst, int Rpad, int Kpad) {
        stage_f16<<<g1((long)Rpad * Kpad), 256, 0, stream>>>(src, R, K, dst, Rpad, Kpad);
    };

    // ---- fold biases ----
    float* enc_b = allocF(2048);
    add2<<<g1(2048), 256, 0, stream>>>(enc_bih, enc_bhh, enc_b, 2048);
    float* lb[6];
    for (int i = 0; i < 6; ++i) {
        lb[i] = allocF(1024);
        add2<<<g1(1024), 256, 0, stream>>>((const float*)d_in[base[i] + 2],
                                           (const float*)d_in[base[i] + 3], lb[i], 1024);
    }

    // ---- stage all weights as padded f16 ----
    _Float16* hwG_h[2]; _Float16* hwL_h[2];
    for (int l = 0; l < 2; ++l) {
        hwG_h[l] = allocH((long)EMBN * EMBP);
        hwL_h[l] = allocH((long)EMBN * EMBP);
        stage(hw_gate_w + (long)l * EMB * EMB, EMB, EMB, hwG_h[l], EMBN, EMBP);
        stage(hw_lin_w  + (long)l * EMB * EMB, EMB, EMB, hwL_h[l], EMBN, EMBP);
    }
    _Float16* encWih_h = allocH((long)2048 * EMBP);
    stage(enc_Wih, 2048, EMB, encWih_h, 2048, EMBP);
    _Float16* encWhh_h = allocH((long)2048 * 512);
    stage(enc_Whh, 2048, 512, encWhh_h, 2048, 512);
    _Float16* Wih_h[6]; _Float16* Whh_h[6];
    for (int i = 0; i < 6; ++i) {
        int Kin = (i < 2) ? 2048 : 512;
        Wih_h[i] = allocH((long)1024 * Kin);
        stage((const float*)d_in[base[i]], 1024, Kin, Wih_h[i], 1024, Kin);
        Whh_h[i] = allocH((long)1024 * 256);
        stage((const float*)d_in[base[i] + 1], 1024, 256, Whh_h[i], 1024, 256);
    }

    // ---- embeddings ----
    float* xC = allocF((long)B_ * LC * EMB);
    float* xQ = allocF((long)B_ * LQ * EMB);
    embed_kernel<<<B_ * LC, 256, 0, stream>>>(Cwid, Ccid, word_mat, char_mat,
                                              dw_w, dw_b, pw_w, pw_b, xC, LC);
    embed_kernel<<<B_ * LQ, 256, 0, stream>>>(Qwid, Qcid, word_mat, char_mat,
                                              dw_w, dw_b, pw_w, pw_b, xQ, LQ);

    // ---- highway x2 (f16 staged activations, padded K) ----
    _Float16* xCh = allocH((long)B_ * LC * EMBP);
    _Float16* xQh = allocH((long)B_ * LQ * EMBP);
    float* gp = allocF((long)B_ * LC * EMB);
    float* lp = allocF((long)B_ * LC * EMB);
    stage(xC, B_ * LC, EMB, xCh, B_ * LC, EMBP);
    stage(xQ, B_ * LQ, EMB, xQh, B_ * LQ, EMBP);
    for (int layer = 0; layer < 2; ++layer) {
        const float* gb  = hw_gate_b + layer * EMB;
        const float* lbv = hw_lin_b + layer * EMB;
        gemm(xCh, EMBP, 0, hwG_h[layer], EMBP, 0, gb,  gp, EMB, 0, B_ * LC, EMB, EMBP, 1, 1);
        gemm(xCh, EMBP, 0, hwL_h[layer], EMBP, 0, lbv, lp, EMB, 0, B_ * LC, EMB, EMBP, 1, 1);
        highway_comb<<<g1((long)B_ * LC * EMB), 256, 0, stream>>>(xC, gp, lp, B_ * LC * EMB);
        stage(xC, B_ * LC, EMB, xCh, B_ * LC, EMBP);
        gemm(xQh, EMBP, 0, hwG_h[layer], EMBP, 0, gb,  gp, EMB, 0, B_ * LQ, EMB, EMBP, 1, 1);
        gemm(xQh, EMBP, 0, hwL_h[layer], EMBP, 0, lbv, lp, EMB, 0, B_ * LQ, EMB, EMBP, 1, 1);
        highway_comb<<<g1((long)B_ * LQ * EMB), 256, 0, stream>>>(xQ, gp, lp, B_ * LQ * EMB);
        stage(xQ, B_ * LQ, EMB, xQh, B_ * LQ, EMBP);
    }

    // ---- encoder LSTM (H=512) ----
    float* preC = allocF((long)B_ * LC * 2048);
    float* preQ = allocF((long)B_ * LQ * 2048);
    gemm(xCh, EMBP, 0, encWih_h, EMBP, 0, enc_b, preC, 2048, 0, B_ * LC, 2048, EMBP, 1, 1);
    gemm(xQh, EMBP, 0, encWih_h, EMBP, 0, enc_b, preQ, 2048, 0, B_ * LQ, 2048, EMBP, 1, 1);
    float* Cenc = allocF((long)B_ * LC * 512);
    float* Qenc = allocF((long)B_ * LQ * 512);
    lstm_scan<<<1, 1024, 192 * 512, stream>>>(preC, encWhh_h, Cenc, 512, LC, 0, 0, 512);
    lstm_scan<<<1, 1024, 192 * 512, stream>>>(preQ, encWhh_h, Qenc, 512, LQ, 0, 0, 512);

    // ---- trilinear attention ----
    float* cterm = allocF(B_ * LC);
    float* qterm = allocF(B_ * LQ);
    rowdot<<<g1((long)B_ * LC * 32), 256, 0, stream>>>(Cenc, 512, att_wc, att_bc, cterm, B_ * LC, 0);
    rowdot<<<g1((long)B_ * LQ * 32), 256, 0, stream>>>(Qenc, 512, att_wq, att_bq, qterm, B_ * LQ, 0);
    _Float16* Qenc_h = allocH((long)B_ * LQ * 512);
    stage(Qenc, B_ * LQ, 512, Qenc_h, B_ * LQ, 512);
    _Float16* Cw_h = allocH((long)B_ * LC * 512);
    cw_mul_f16<<<g1((long)B_ * LC * 512), 256, 0, stream>>>(Cenc, att_wcq, Cw_h, B_ * LC * 512, 512);
    float* sbuf = allocF((long)B_ * LC * LQ);
    gemm(Cw_h, 512, (long)LC * 512, Qenc_h, 512, (long)LQ * 512, nullptr,
         sbuf, LQ, (long)LC * LQ, LC, LQ, 512, 1, B_);
    float* smax = allocF(B_ * LC);
    _Float16* a_h = allocH((long)B_ * LC * LQ);
    attn_softmax<<<g1((long)B_ * LC * 32), 256, 0, stream>>>(sbuf, cterm, qterm, att_bcq, a_h, smax);
    float* c2q = allocF((long)B_ * LC * 512);
    gemm(a_h, LQ, (long)LC * LQ, Qenc_h, 512, (long)LQ * 512, nullptr,
         c2q, 512, (long)LC * 512, LC, 512, LQ, 0, B_);
    float* q2c = allocF(B_ * 512);
    q2c_kernel<<<B_, 256, 0, stream>>>(smax, Cenc, q2c);
    float* gmat = allocF((long)B_ * LC * 2048);
    _Float16* gmat_h = allocH((long)B_ * LC * 2048);
    build_g<<<g1((long)B_ * LC * 2048), 256, 0, stream>>>(Cenc, c2q, q2c, gmat, gmat_h, B_ * LC * 2048);

    // ---- three stacked BiLSTMs (H=256) ----
    float* preA = allocF((long)B_ * LC * 1024);
    float* preB = allocF((long)B_ * LC * 1024);
    float* m1o  = allocF((long)B_ * LC * 512);
    float* mo   = allocF((long)B_ * LC * 512);
    float* m2o  = allocF((long)B_ * LC * 512);
    _Float16* m1o_h = allocH((long)B_ * LC * 512);
    _Float16* mo_h  = allocH((long)B_ * LC * 512);

    gemm(gmat_h, 2048, 0, Wih_h[0], 2048, 0, lb[0], preA, 1024, 0, B_ * LC, 1024, 2048, 1, 1);
    gemm(gmat_h, 2048, 0, Wih_h[1], 2048, 0, lb[1], preB, 1024, 0, B_ * LC, 1024, 2048, 1, 1);
    lstm_scan<<<1, 1024, 192 * 256, stream>>>(preA, Whh_h[0], m1o, 256, LC, 0,   0, 512);
    lstm_scan<<<1, 1024, 192 * 256, stream>>>(preB, Whh_h[1], m1o, 256, LC, 1, 256, 512);
    stage(m1o, B_ * LC, 512, m1o_h, B_ * LC, 512);

    gemm(m1o_h, 512, 0, Wih_h[2], 512, 0, lb[2], preA, 1024, 0, B_ * LC, 1024, 512, 1, 1);
    gemm(m1o_h, 512, 0, Wih_h[3], 512, 0, lb[3], preB, 1024, 0, B_ * LC, 1024, 512, 1, 1);
    lstm_scan<<<1, 1024, 192 * 256, stream>>>(preA, Whh_h[2], mo, 256, LC, 0,   0, 512);
    lstm_scan<<<1, 1024, 192 * 256, stream>>>(preB, Whh_h[3], mo, 256, LC, 1, 256, 512);
    stage(mo, B_ * LC, 512, mo_h, B_ * LC, 512);

    gemm(mo_h, 512, 0, Wih_h[4], 512, 0, lb[4], preA, 1024, 0, B_ * LC, 1024, 512, 1, 1);
    gemm(mo_h, 512, 0, Wih_h[5], 512, 0, lb[5], preB, 1024, 0, B_ * LC, 1024, 512, 1, 1);
    lstm_scan<<<1, 1024, 192 * 256, stream>>>(preA, Whh_h[4], m2o, 256, LC, 0,   0, 512);
    lstm_scan<<<1, 1024, 192 * 256, stream>>>(preB, Whh_h[5], m2o, 256, LC, 1, 256, 512);

    // ---- output projections + masked log-softmax ----
    float* p1 = allocF(B_ * LC);
    float* p2 = allocF(B_ * LC);
    rowdot<<<g1((long)B_ * LC * 32), 256, 0, stream>>>(gmat, 2048, (const float*)d_in[48],
                                                       (const float*)d_in[49], p1, B_ * LC, 0);
    rowdot<<<g1((long)B_ * LC * 32), 256, 0, stream>>>(mo,   512, (const float*)d_in[50],
                                                       (const float*)d_in[51], p1, B_ * LC, 1);
    rowdot<<<g1((long)B_ * LC * 32), 256, 0, stream>>>(gmat, 2048, (const float*)d_in[52],
                                                       (const float*)d_in[53], p2, B_ * LC, 0);
    rowdot<<<g1((long)B_ * LC * 32), 256, 0, stream>>>(m2o,  512, (const float*)d_in[54],
                                                       (const float*)d_in[55], p2, B_ * LC, 1);

    mask_logsoftmax<<<B_, 256, 0, stream>>>(p1, Cwid, (float*)d_out);
    mask_logsoftmax<<<B_, 256, 0, stream>>>(p2, Cwid, (float*)d_out + B_ * LC);
}
